// Enc2DecWithAttention_45122926412147
// MI455X (gfx1250) — matrix operand order repaired and checked by reading, not compiled, not run on hardware
//
#include <hip/hip_runtime.h>

#define DEV __device__ __forceinline__

typedef __attribute__((ext_vector_type(16))) _Float16 v16h;
typedef __attribute__((ext_vector_type(8)))  float    v8f;
typedef __attribute__((ext_vector_type(4)))  unsigned int u32x4;
typedef __attribute__((ext_vector_type(4)))  int i32x4;
typedef __attribute__((ext_vector_type(8)))  int i32x8;

#if defined(__has_builtin)
#  if __has_builtin(__builtin_amdgcn_tensor_load_to_lds) && __has_builtin(__builtin_amdgcn_s_wait_tensorcnt)
#    define HAVE_TDM 1
#  endif
#endif
#ifndef HAVE_TDM
#  define HAVE_TDM 0
#endif

// Problem sizes (fixed by the reference)
constexpr int BATCH = 64, SEQ = 64, LOUT = 64, HD = 256, NVIN = 10000, NVOUT = 10000;
constexpr int H3 = 3 * HD;           // 768
constexpr int HS = HD + 4;           // padded LDS row stride (floats) -> conflict-free
constexpr int ATS = HD + 8;          // padded f16 LDS stride (264)
constexpr int KFULL = (NVIN / 32) * 32;   // 9984; tail is exactly 16

DEV v8f zero8() { v8f z;
#pragma unroll
  for (int i = 0; i < 8; ++i) z[i] = 0.f; return z; }

DEV v8f wmma16(v16h a, v16h b, v8f c) {
  return __builtin_amdgcn_wmma_f32_16x16x32_f16(false, a, false, b, (short)0, c, false, false);
}

DEV float sigm(float x) { return 1.f / (1.f + expf(-x)); }

// ---- fragment builders (ISA 7.12.2 layouts, wave32) ----
// A 16x32 f16 (MxK): lane m=l&15, kh=l>>4; e<8 -> K=kh*8+e ; e>=8 -> K=16+kh*8+(e-8)
DEV v16h afrag_f32_fast(const float* rowptr) {          // rowptr = row start + kbase (16B aligned)
  int kh = (threadIdx.x >> 4) & 1;
  const float* p = rowptr + kh * 8;
  float4 x0 = *(const float4*)(p);
  float4 x1 = *(const float4*)(p + 4);
  float4 y0 = *(const float4*)(p + 16);
  float4 y1 = *(const float4*)(p + 20);
  v16h f;
  f[0]=(_Float16)x0.x; f[1]=(_Float16)x0.y; f[2]=(_Float16)x0.z; f[3]=(_Float16)x0.w;
  f[4]=(_Float16)x1.x; f[5]=(_Float16)x1.y; f[6]=(_Float16)x1.z; f[7]=(_Float16)x1.w;
  f[8]=(_Float16)y0.x; f[9]=(_Float16)y0.y; f[10]=(_Float16)y0.z; f[11]=(_Float16)y0.w;
  f[12]=(_Float16)y1.x; f[13]=(_Float16)y1.y; f[14]=(_Float16)y1.z; f[15]=(_Float16)y1.w;
  return f;
}
// K-tail of exactly 16: valid K<16 -> elements 0..7 (K=kh*8+e) valid, 8..15 zero
DEV v16h afrag_f32_tail16(const float* rowptr) {
  int kh = (threadIdx.x >> 4) & 1;
  const float* p = rowptr + kh * 8;
  float4 x0 = *(const float4*)(p);
  float4 x1 = *(const float4*)(p + 4);
  v16h f;
  f[0]=(_Float16)x0.x; f[1]=(_Float16)x0.y; f[2]=(_Float16)x0.z; f[3]=(_Float16)x0.w;
  f[4]=(_Float16)x1.x; f[5]=(_Float16)x1.y; f[6]=(_Float16)x1.z; f[7]=(_Float16)x1.w;
#pragma unroll
  for (int e = 8; e < 16; ++e) f[e] = (_Float16)0.f;
  return f;
}
DEV v16h afrag_f16_fast(const _Float16* rowptr) {
  int kh = (threadIdx.x >> 4) & 1;
  union { uint4 u[2]; v16h v; } t;
  t.u[0] = *(const uint4*)(rowptr + kh * 8);
  t.u[1] = *(const uint4*)(rowptr + 16 + kh * 8);
  return t.v;
}
// B 32x16 f16 (KxN): lane n=l&15, kh=l>>4; e<8 -> K=kh*8+e ; e>=8 -> K=16+kh*8+(e-8)
DEV v16h bfrag_f16(const _Float16* W, int ld, int rowbase, int kbase) {
  int lane = threadIdx.x & 31;
  const _Float16* p = W + (size_t)(rowbase + (lane & 15)) * ld + kbase + (lane >> 4) * 8;
  union { uint4 u[2]; v16h v; } t;
  t.u[0] = *(const uint4*)(p);
  t.u[1] = *(const uint4*)(p + 16);
  return t.v;
}
// K-tail of exactly 16: elements 0..7 (K=kh*8+e) valid, 8..15 zero
DEV v16h bfrag_f16_tail16(const _Float16* W, int ld, int rowbase, int kbase) {
  int lane = threadIdx.x & 31;
  union { uint4 u[2]; v16h v; } t;
  uint4 z; z.x = 0; z.y = 0; z.z = 0; z.w = 0;
  t.u[0] = z; t.u[1] = z;
  {
    const _Float16* p = W + (size_t)(rowbase + (lane & 15)) * ld + kbase + (lane >> 4) * 8;
    t.u[0] = *(const uint4*)(p);
  }
  return t.v;
}

// ---------------- fp32 -> f16 conversion (strided sub-block) ----------------
__global__ void k_cvt(const float* __restrict__ src, _Float16* __restrict__ dst,
                      int rows, int src_ld, int cols) {
  long total = (long)rows * cols;
  long stride = (long)gridDim.x * blockDim.x;
  for (long idx = (long)blockIdx.x * blockDim.x + threadIdx.x; idx < total; idx += stride) {
    int r = (int)(idx / cols), c = (int)(idx - (long)r * cols);
    dst[idx] = (_Float16)src[(size_t)r * src_ld + c];
  }
}

// ---------------- K1: GI[s*64+b, 0:768] = X[b,s,:] @ enc_Wih^T + bih ----------------
// grid (32, 12), block 256 (8 waves, 4x2), wave tile 32x32, branch-free K loop + peeled tail
__global__ __launch_bounds__(256) void k1_gi(const float* __restrict__ X,
                                             const _Float16* __restrict__ Wih,
                                             const float* __restrict__ bih,
                                             float* __restrict__ GI) {
  int w = threadIdx.x >> 5, lane = threadIdx.x & 31;
  int wm = w & 3, wn = w >> 2;
  int mbase = blockIdx.x * 128 + wm * 32;
  int nbase = blockIdx.y * 64 + wn * 32;
  int m = lane & 15, lh = lane >> 4, nn = lane & 15;
  int mg0 = mbase + m, mg1 = mbase + 16 + m;
  const float* arow0 = X + (size_t)((mg0 & 63) * 64 + (mg0 >> 6)) * NVIN;   // X[b][s][:]
  const float* arow1 = X + (size_t)((mg1 & 63) * 64 + (mg1 >> 6)) * NVIN;
  v8f acc[2][2] = {{zero8(), zero8()}, {zero8(), zero8()}};
  for (int kb = 0; kb < KFULL; kb += 32) {
    __builtin_prefetch(arow0 + kb + 128, 0, 3);
    v16h a0 = afrag_f32_fast(arow0 + kb);
    v16h a1 = afrag_f32_fast(arow1 + kb);
    v16h b0 = bfrag_f16(Wih, NVIN, nbase, kb);
    v16h b1 = bfrag_f16(Wih, NVIN, nbase + 16, kb);
    acc[0][0] = wmma16(a0, b0, acc[0][0]);
    acc[0][1] = wmma16(a0, b1, acc[0][1]);
    acc[1][0] = wmma16(a1, b0, acc[1][0]);
    acc[1][1] = wmma16(a1, b1, acc[1][1]);
  }
  { // peeled tail: K remainder = 16
    v16h a0 = afrag_f32_tail16(arow0 + KFULL);
    v16h a1 = afrag_f32_tail16(arow1 + KFULL);
    v16h b0 = bfrag_f16_tail16(Wih, NVIN, nbase, KFULL);
    v16h b1 = bfrag_f16_tail16(Wih, NVIN, nbase + 16, KFULL);
    acc[0][0] = wmma16(a0, b0, acc[0][0]);
    acc[0][1] = wmma16(a0, b1, acc[0][1]);
    acc[1][0] = wmma16(a1, b0, acc[1][0]);
    acc[1][1] = wmma16(a1, b1, acc[1][1]);
  }
#pragma unroll
  for (int i = 0; i < 2; ++i)
#pragma unroll
    for (int j = 0; j < 2; ++j)
#pragma unroll
      for (int r = 0; r < 8; ++r) {
        int mm = mbase + i * 16 + r + lh * 8;
        int nc = nbase + j * 16 + nn;
        GI[mm * H3 + nc] = acc[i][j][r] + bih[nc];
      }
}

// ---------------- K2: encoder recurrence (persistent, 1 WG x 1024 thr = 32 waves) ----------------
// per step: gh = h @ Whh^T (WMMA, 64 units of 16x16x3-gates), fuse gates, ping-pong h in LDS
__global__ __launch_bounds__(1024) void k2_enc_scan(const _Float16* __restrict__ Whh,
                                                    const float* __restrict__ bhh,
                                                    const float* __restrict__ GI,
                                                    float* __restrict__ enc_h) {
  extern __shared__ float smem[];
  float* h0buf = smem;
  float* h1buf = smem + BATCH * HS;
  int tid = threadIdx.x, w = tid >> 5, lane = tid & 31;
  int m = lane & 15, lh = lane >> 4, nn = lane & 15;
  for (int idx = tid; idx < BATCH * HS; idx += 1024) h0buf[idx] = 0.f;
  __syncthreads();
  for (int s = 0; s < SEQ; ++s) {
    const float* hr = (s & 1) ? h1buf : h0buf;
    float* hw = (s & 1) ? h0buf : h1buf;
#pragma unroll
    for (int u2 = 0; u2 < 2; ++u2) {
      int u = w + u2 * 32;                 // 64 units = 4 mtiles x 16 itiles
      int m0 = (u >> 4) * 16, i0 = (u & 15) * 16;
      v8f aR = zero8(), aZ = zero8(), aN = zero8();
      const float* hrow = hr + (m0 + m) * HS;
#pragma unroll
      for (int kc = 0; kc < 8; ++kc) {
        int kb = kc * 32;
        v16h a = afrag_f32_fast(hrow + kb);
        aR = wmma16(a, bfrag_f16(Whh, HD, i0, kb), aR);
        aZ = wmma16(a, bfrag_f16(Whh, HD, HD + i0, kb), aZ);
        aN = wmma16(a, bfrag_f16(Whh, HD, 2 * HD + i0, kb), aN);
      }
#pragma unroll
      for (int r = 0; r < 8; ++r) {
        int b = m0 + r + lh * 8;
        int i = i0 + nn;
        const float* gi = GI + (s * 64 + b) * H3;      // bih already folded in
        float rr = sigm(gi[i] + aR[r] + bhh[i]);
        float zz = sigm(gi[HD + i] + aZ[r] + bhh[HD + i]);
        float nv = tanhf(gi[2 * HD + i] + rr * (aN[r] + bhh[2 * HD + i]));
        float hnew = (1.f - zz) * nv + zz * hr[b * HS + i];
        hw[b * HS + i] = hnew;
        enc_h[((s * 64 + b) << 8) + i] = hnew;
      }
    }
    __syncthreads();
  }
}

// ---------------- K2b: enc_scoreT[b,s] = enc_h[s,b,:] . w_enc ----------------
__global__ __launch_bounds__(256) void k2b_score(const float* __restrict__ enc_h,
                                                 const float* __restrict__ attw,
                                                 float* __restrict__ scoreT) {
  int id = blockIdx.x * 8 + (threadIdx.x >> 5);
  int lane = threadIdx.x & 31;
  int s = id >> 6, b = id & 63;
  float p = 0.f;
  for (int k = lane; k < HD; k += 32)
    p += enc_h[((s * 64 + b) << 8) + k] * attw[HD + k];
  for (int msk = 16; msk; msk >>= 1) p += __shfl_xor(p, msk, 32);
  if (lane == 0) scoreT[(b << 6) + s] = p;
}

// ---------------- K3: decoder recurrence (persistent, 1 WG x 1024) ----------------
// attention softmax + ctx + dual-A WMMA (h x Whh, ctx x Wih) + one-hot gather + GRU fuse
__global__ __launch_bounds__(1024) void k3_dec_scan(const float* __restrict__ enc_h,
                                                    const float* __restrict__ scoreT,
                                                    const float* __restrict__ attw,
                                                    const float* __restrict__ attb,
                                                    const _Float16* __restrict__ Whh,
                                                    const _Float16* __restrict__ WihH,
                                                    const float* __restrict__ WihFull,
                                                    const float* __restrict__ bih,
                                                    const float* __restrict__ bhh,
                                                    const int* __restrict__ gt,
                                                    _Float16* __restrict__ H2h) {
  extern __shared__ float smem[];
  float* h0buf = smem;
  float* h1buf = smem + BATCH * HS;
  float* ctx = smem + 2 * BATCH * HS;                  // [64][HS]
  float* wts = ctx + BATCH * HS;                       // [64][68] as [s*68+b]
  float* hdot = wts + 64 * 68;                         // [64]
  int tid = threadIdx.x, w = tid >> 5, lane = tid & 31;
  int m = lane & 15, lh = lane >> 4, nn = lane & 15;
  for (int idx = tid; idx < BATCH * HD; idx += 1024) { // h0 = enc_h[S-1]
    int b = idx >> 8, i = idx & 255;
    h0buf[b * HS + i] = enc_h[(((SEQ - 1) * 64 + b) << 8) + i];
  }
  __syncthreads();
  float attB = attb[0];
  for (int l = 0; l < LOUT; ++l) {
    const float* hr = (l & 1) ? h1buf : h0buf;
    float* hw = (l & 1) ? h0buf : h1buf;
    // --- hdot[b] = h . w_dec ---
    {
      int bb = w;                                      // waves 0..31 -> b, b+32
#pragma unroll
      for (int half = 0; half < 2; ++half, bb += 32) {
        float p = 0.f;
        for (int k = lane; k < HD; k += 32) p += hr[bb * HS + k] * attw[k];
        for (int msk = 16; msk; msk >>= 1) p += __shfl_xor(p, msk, 32);
        if (lane == 0) hdot[bb] = p;
      }
    }
    __syncthreads();
    // --- softmax over s -> wts ---
    {
      int bb = w;
#pragma unroll
      for (int half = 0; half < 2; ++half, bb += 32) {
        float hb = hdot[bb] + attB;
        float v0 = fmaxf(scoreT[(bb << 6) + lane] + hb, 0.f);
        float v1 = fmaxf(scoreT[(bb << 6) + lane + 32] + hb, 0.f);
        float mx = fmaxf(v0, v1);
        for (int msk = 16; msk; msk >>= 1) mx = fmaxf(mx, __shfl_xor(mx, msk, 32));
        float e0 = expf(v0 - mx), e1 = expf(v1 - mx);
        float sm = e0 + e1;
        for (int msk = 16; msk; msk >>= 1) sm += __shfl_xor(sm, msk, 32);
        float inv = 1.f / sm;
        wts[lane * 68 + bb] = e0 * inv;
        wts[(lane + 32) * 68 + bb] = e1 * inv;
      }
    }
    __syncthreads();
    // --- ctx[b,i] = sum_s wts[s,b] * enc_h[s,b,i] ---
    for (int idx = tid; idx < BATCH * HD; idx += 1024) {
      int b = idx >> 8, i = idx & 255;
      float acc = 0.f;
      for (int s = 0; s < SEQ; ++s)
        acc += wts[s * 68 + b] * enc_h[((s * 64 + b) << 8) + i];
      ctx[b * HS + i] = acc;
    }
    __syncthreads();
    // --- dual-source WMMA + GRU fuse ---
#pragma unroll
    for (int u2 = 0; u2 < 2; ++u2) {
      int u = w + u2 * 32;
      int m0 = (u >> 4) * 16, i0 = (u & 15) * 16;
      v8f aR = zero8(), aZ = zero8(), aGN = zero8(), aIN = zero8();
      const float* hrow = hr + (m0 + m) * HS;
      const float* crow = ctx + (m0 + m) * HS;
#pragma unroll
      for (int kc = 0; kc < 8; ++kc) {
        int kb = kc * 32;
        v16h aH = afrag_f32_fast(hrow + kb);
        v16h aC = afrag_f32_fast(crow + kb);
        aR = wmma16(aH, bfrag_f16(Whh, HD, i0, kb), aR);
        aR = wmma16(aC, bfrag_f16(WihH, HD, i0, kb), aR);
        aZ = wmma16(aH, bfrag_f16(Whh, HD, HD + i0, kb), aZ);
        aZ = wmma16(aC, bfrag_f16(WihH, HD, HD + i0, kb), aZ);
        aGN = wmma16(aH, bfrag_f16(Whh, HD, 2 * HD + i0, kb), aGN);
        aIN = wmma16(aC, bfrag_f16(WihH, HD, 2 * HD + i0, kb), aIN);
      }
#pragma unroll
      for (int r = 0; r < 8; ++r) {
        int b = m0 + r + lh * 8;
        int i = i0 + nn;
        int g = (l > 0) ? gt[(b << 6) + l - 1] : -1;   // teacher forcing (one-hot gather)
        float ohr = 0.f, ohz = 0.f, ohn = 0.f;
        if (g >= 0) {
          ohr = WihFull[(size_t)i * (HD + NVOUT) + HD + g];
          ohz = WihFull[(size_t)(HD + i) * (HD + NVOUT) + HD + g];
          ohn = WihFull[(size_t)(2 * HD + i) * (HD + NVOUT) + HD + g];
        }
        float rr = sigm(aR[r] + bih[i] + bhh[i] + ohr);
        float zz = sigm(aZ[r] + bih[HD + i] + bhh[HD + i] + ohz);
        float inn = aIN[r] + bih[2 * HD + i] + ohn;
        float hn = aGN[r] + bhh[2 * HD + i];
        float nv = tanhf(inn + rr * hn);
        float hnew = (1.f - zz) * nv + zz * hr[b * HS + i];
        hw[b * HS + i] = hnew;
        H2h[((l * 64 + b) << 8) + i] = (_Float16)hnew;
      }
    }
    __syncthreads();
  }
}

// ---------------- K4: logits GEMM [4096x10000] + online softmax stats ----------------
// 1 WG per 16-row group; A tile staged via Tensor Data Mover (async DMA to LDS with
// hardware row padding), raw z written into d_out (as scratch), row max/sumexp to ws
__global__ __launch_bounds__(256) void k4_out(const _Float16* __restrict__ H2h,
                                              const _Float16* __restrict__ Wo,
                                              const float* __restrict__ bo,
                                              float* __restrict__ out,
                                              float* __restrict__ rowM,
                                              float* __restrict__ rowS) {
  __shared__ _Float16 At[16 * ATS];
  __shared__ float smM[8][16], smS[8][16];
  int tid = threadIdx.x, w = tid >> 5, lane = tid & 31;
  int g = blockIdx.x;
#if HAVE_TDM
  if (w == 0) {
    // D# descriptor: 2D tensor [4096 x 256] f16, tile [16 x 256] at row g*16,
    // LDS dest = At with 4-DWORD pad every 128 DWORDs (-> 264-f16 row stride)
    unsigned ldsa = (unsigned)(uintptr_t)(&At[0]);             // low 32 bits = LDS byte addr
    unsigned long long ga = (unsigned long long)(uintptr_t)(H2h + (size_t)g * 16 * HD);
    u32x4 g0;
    g0[0] = 1u;                                                // count=1 (valid), user mode
    g0[1] = ldsa;                                              // lds_addr
    g0[2] = (unsigned)(ga & 0xFFFFFFFFu);                      // global_addr[31:0]
    g0[3] = (unsigned)((ga >> 32) & 0x1FFFFFFu) | (2u << 30);  // global_addr[56:32] | type=2
    i32x8 g1;
    g1[0] = (int)((1u << 16) | (1u << 20) | (6u << 22) | (3u << 25)); // 2B elems, pad 4DW/128DW
    g1[1] = (int)(256u << 16);       // tensor_dim0 = 256
    g1[2] = (int)(4096u << 16);      // tensor_dim1 = 4096
    g1[3] = (int)(256u << 16);       // tile_dim0 = 256
    g1[4] = (int)16;                 // tile_dim1 = 16
    g1[5] = (int)256;                // tensor_dim0_stride = 256
    g1[6] = (int)(256u << 16);       // tensor_dim1_stride = 256
    g1[7] = 0;
    i32x4 gz4; gz4[0] = 0; gz4[1] = 0; gz4[2] = 0; gz4[3] = 0;
#if __clang_major__ >= 23
    i32x8 gz8;
#pragma unroll
    for (int i = 0; i < 8; ++i) gz8[i] = 0;
    __builtin_amdgcn_tensor_load_to_lds(g0, g1, gz4, gz4, gz8, 0);
#else
    __builtin_amdgcn_tensor_load_to_lds(g0, g1, gz4, gz4, 0);
#endif
    __builtin_amdgcn_s_wait_tensorcnt(0);
  }
  __syncthreads();
#else
  for (int idx = tid; idx < 16 * HD; idx += 256) {
    int r = idx >> 8, c = idx & 255;
    At[r * ATS + c] = H2h[((g * 16 + r) << 8) + c];
  }
  __syncthreads();
#endif
  int m = lane & 15, lh = lane >> 4, nn = lane & 15;
  v16h a[8];
#pragma unroll
  for (int kc = 0; kc < 8; ++kc) a[kc] = afrag_f16_fast(&At[m * ATS + kc * 32]);
  float runm[8], runs[8];
#pragma unroll
  for (int r = 0; r < 8; ++r) { runm[r] = -__builtin_inff(); runs[r] = 0.f; }
  for (int c = 0; c < 79; ++c) {                 // 78 full 128-wide chunks + 16-wide tail
    int n0 = c * 128 + w * 16;
    if (n0 >= NVOUT) continue;                   // uniform per wave
    v8f acc = zero8();
#pragma unroll
    for (int kc = 0; kc < 8; ++kc)
      acc = wmma16(a[kc], bfrag_f16(Wo, HD, n0, kc * 32), acc);
#pragma unroll
    for (int r = 0; r < 8; ++r) {
      int n = n0 + nn;
      float z = acc[r] + bo[n];
      int mg = g * 16 + r + lh * 8;              // row = l*64 + b
      out[(((mg & 63) << 6) + (mg >> 6)) * NVOUT + n] = z;   // d_out[b][l][v] raw
      float nm = fmaxf(runm[r], z);
      runs[r] = runs[r] * expf(runm[r] - nm) + expf(z - nm);
      runm[r] = nm;
    }
  }
  for (int msk = 1; msk < 16; msk <<= 1) {       // combine 16 lanes of same row-half
#pragma unroll
    for (int r = 0; r < 8; ++r) {
      float om = __shfl_xor(runm[r], msk, 32);
      float os = __shfl_xor(runs[r], msk, 32);
      float nm = fmaxf(runm[r], om);
      runs[r] = runs[r] * expf(runm[r] - nm) + os * expf(om - nm);
      runm[r] = nm;
    }
  }
  if (nn == 0) {                                  // lanes 0 and 16
#pragma unroll
    for (int r = 0; r < 8; ++r) { smM[w][lh * 8 + r] = runm[r]; smS[w][lh * 8 + r] = runs[r]; }
  }
  __syncthreads();
  if (tid < 16) {
    float mM = -__builtin_inff(), sS = 0.f;
    for (int ww = 0; ww < 8; ++ww) {
      float om = smM[ww][tid], os = smS[ww][tid];
      float nm = fmaxf(mM, om);
      sS = sS * expf(mM - nm) + os * expf(om - nm);
      mM = nm;
    }
    rowM[g * 16 + tid] = mM;
    rowS[g * 16 + tid] = sS;
  }
}

// ---------------- K5: logp = z - max - log(sumexp) (streaming finalize) ----------------
__global__ void k5_fin(float* __restrict__ out, const float* __restrict__ rowM,
                       const float* __restrict__ rowS) {
  int stride = gridDim.x * blockDim.x;
  const int total = BATCH * LOUT * NVOUT;
  for (int idx = blockIdx.x * blockDim.x + threadIdx.x; idx < total; idx += stride) {
    int t = idx / NVOUT;                          // t = b*64 + l
    int mrow = ((t & 63) << 6) + (t >> 6);        // l*64 + b
    out[idx] = out[idx] - rowM[mrow] - logf(rowS[mrow]);
  }
}

extern "C" void kernel_launch(void* const* d_in, const int* in_sizes, int n_in,
                              void* d_out, int out_size, void* d_ws, size_t ws_size,
                              hipStream_t stream) {
  const float* X       = (const float*)d_in[0];
  const int*   gt      = (const int*)d_in[1];
  const float* eWih    = (const float*)d_in[2];
  const float* eWhh    = (const float*)d_in[3];
  const float* eBih    = (const float*)d_in[4];
  const float* eBhh    = (const float*)d_in[5];
  const float* attw    = (const float*)d_in[6];
  const float* attb    = (const float*)d_in[7];
  const float* dWih    = (const float*)d_in[8];
  const float* dWhh    = (const float*)d_in[9];
  const float* dBih    = (const float*)d_in[10];
  const float* dBhh    = (const float*)d_in[11];
  const float* outW    = (const float*)d_in[12];
  const float* outB    = (const float*)d_in[13];
  float* out = (float*)d_out;

  char* ws = (char*)d_ws;
  size_t o = 0;
  auto take = [&](size_t bytes) { char* p = ws + o; o = (o + bytes + 255) & ~(size_t)255; return p; };
  _Float16* eWih_h = (_Float16*)take((size_t)H3 * NVIN * 2);
  _Float16* eWhh_h = (_Float16*)take((size_t)H3 * HD * 2);
  _Float16* dWhh_h = (_Float16*)take((size_t)H3 * HD * 2);
  _Float16* dWihH_h = (_Float16*)take((size_t)H3 * HD * 2);
  _Float16* outW_h = (_Float16*)take((size_t)NVOUT * HD * 2);
  float* GI      = (float*)take((size_t)SEQ * BATCH * H3 * 4);
  float* enc_h   = (float*)take((size_t)SEQ * BATCH * HD * 4);
  float* scoreT  = (float*)take((size_t)BATCH * SEQ * 4);
  _Float16* H2h  = (_Float16*)take((size_t)LOUT * BATCH * HD * 2);
  float* rowM    = (float*)take((size_t)LOUT * BATCH * 4);
  float* rowS    = (float*)take((size_t)LOUT * BATCH * 4);
  (void)ws_size; (void)n_in; (void)in_sizes; (void)out_size;

  // weight conversions to f16
  k_cvt<<<2048, 256, 0, stream>>>(eWih, eWih_h, H3, NVIN, NVIN);
  k_cvt<<<256, 256, 0, stream>>>(eWhh, eWhh_h, H3, HD, HD);
  k_cvt<<<256, 256, 0, stream>>>(dWhh, dWhh_h, H3, HD, HD);
  k_cvt<<<256, 256, 0, stream>>>(dWih, dWihH_h, H3, HD + NVOUT, HD);  // first H cols only
  k_cvt<<<1024, 256, 0, stream>>>(outW, outW_h, NVOUT, HD, HD);

  // big hoisted GEMM: encoder input gates for all timesteps
  k1_gi<<<dim3(32, 12), 256, 0, stream>>>(X, eWih_h, eBih, GI);

  // persistent encoder recurrence (32 waves, ping-pong h in LDS)
  k2_enc_scan<<<1, 1024, 2 * BATCH * HS * sizeof(float), stream>>>(eWhh_h, eBhh, GI, enc_h);

  // precomputed attention encoder scores
  k2b_score<<<512, 256, 0, stream>>>(enc_h, attw, scoreT);

  // persistent decoder recurrence with fused attention
  size_t smem3 = (size_t)(2 * BATCH * HS + BATCH * HS + 64 * 68 + 64) * sizeof(float);
  k3_dec_scan<<<1, 1024, smem3, stream>>>(enc_h, scoreT, attw, attb, dWhh_h, dWihH_h,
                                          dWih, dBih, dBhh, gt, H2h);

  // output GEMM + online log-softmax stats (raw z staged in d_out)
  k4_out<<<LOUT * BATCH / 16, 256, 0, stream>>>(H2h, outW_h, outB, out, rowM, rowS);

  // finalize log-softmax
  k5_fin<<<8192, 256, 0, stream>>>(out, rowM, rowS);
}
